// PruningUnit_20091857011344
// MI455X (gfx1250) — compile-verified, matrix operand order checked by reading, not run on hardware
//
#include <hip/hip_runtime.h>
#include <hip/hip_bf16.h>
#include <math.h>

// ---------------------------------------------------------------------------
// Problem constants (fixed by the reference)
// ---------------------------------------------------------------------------
constexpr int N   = 1536;
constexpr int E   = 24576;
constexpr int E2  = E + N;          // edges + self loops
constexpr int HD  = 128;
constexpr int D   = 1280;
constexpr float NEGV = -1000000000.0f;

typedef __bf16 bf16;
typedef __attribute__((ext_vector_type(8)))  __bf16 v8bf;
typedef __attribute__((ext_vector_type(16))) __bf16 v16bf;
typedef __attribute__((ext_vector_type(8)))  float  v8f;

// ---------------------------------------------------------------------------
// Helpers
// ---------------------------------------------------------------------------
__device__ __forceinline__ unsigned fflip(float f) {
    unsigned u = __float_as_uint(f);
    return u ^ ((unsigned)((int)u >> 31) | 0x80000000u);
}
__device__ __forceinline__ float funflip(unsigned v) {
    unsigned x = (~(unsigned)((int)v >> 31)) | 0x80000000u;
    return __uint_as_float(v ^ x);
}
__device__ __forceinline__ void edge_st(const int* __restrict__ ei, int j, int& s, int& t) {
    if (j < E) { s = ei[j]; t = ei[E + j]; }
    else       { s = j - E; t = j - E; }
}
__device__ __forceinline__ float leaky(float v) { return v > 0.0f ? v : 0.2f * v; }

// Load one 16x32 bf16 fragment (A layout; B^T uses the mirrored identical layout).
// Lane l (r = l&15, h = l>>4): K = h*8 + [0..7]  and  16 + h*8 + [0..7]
__device__ __forceinline__ v16bf load_frag(const bf16* __restrict__ p) {
    v8bf lo = *reinterpret_cast<const v8bf*>(p);
    v8bf hi = *reinterpret_cast<const v8bf*>(p + 16);
    return __builtin_shufflevector(lo, hi, 0,1,2,3,4,5,6,7,8,9,10,11,12,13,14,15);
}

// ---------------------------------------------------------------------------
// bf16 WMMA GEMM:  C[M x Ncols] = A[M x K] * B[K x Ncols],  B given as
// Bt (row-major B^T, i.e. Bt[n][k]).  Block = 128 threads = 4 waves,
// each wave computes a 16x64 strip (1 A frag -> 4 WMMAs / k-step).
// M, Ncols multiples of 64; K multiple of 32; lda/ldb multiples of 8.
// ---------------------------------------------------------------------------
__global__ __launch_bounds__(128)
void k_gemm_bf16(const bf16* __restrict__ A, int lda,
                 const bf16* __restrict__ Bt, int ldb,
                 float* __restrict__ C, int ldc, int K)
{
    const int lane = threadIdx.x & 31;
    const int wave = threadIdx.x >> 5;
    const int rr = lane & 15;
    const int hh = lane >> 4;
    const int m0 = blockIdx.y * 64 + wave * 16;
    const int n0 = blockIdx.x * 64;

    const bf16* ap  = A  + (size_t)(m0 + rr) * lda + hh * 8;
    const bf16* bp  = Bt + (size_t)(n0 + rr) * ldb + hh * 8;
    const size_t bstep = (size_t)16 * ldb;

    v8f acc0 = {}, acc1 = {}, acc2 = {}, acc3 = {};

    for (int k = 0; k < K; k += 32) {
        // L2-resident working set; hint next tiles anyway (speculative is safe).
        __builtin_prefetch(ap + k + 128, 0, 1);
        __builtin_prefetch(bp + k + 128, 0, 1);

        v16bf af = load_frag(ap + k);
        v16bf b0 = load_frag(bp + k);
        v16bf b1 = load_frag(bp + bstep + k);
        v16bf b2 = load_frag(bp + 2 * bstep + k);
        v16bf b3 = load_frag(bp + 3 * bstep + k);

        acc0 = __builtin_amdgcn_wmma_f32_16x16x32_bf16(false, af, false, b0, (short)0, acc0, false, false);
        acc1 = __builtin_amdgcn_wmma_f32_16x16x32_bf16(false, af, false, b1, (short)0, acc1, false, false);
        acc2 = __builtin_amdgcn_wmma_f32_16x16x32_bf16(false, af, false, b2, (short)0, acc2, false, false);
        acc3 = __builtin_amdgcn_wmma_f32_16x16x32_bf16(false, af, false, b3, (short)0, acc3, false, false);
    }

    // C/D layout: lane l -> col n0 + rr (+16*nn); VGPR j -> row m0 + hh*8 + j
    float* c0 = C + (size_t)(m0 + hh * 8) * ldc + n0 + rr;
#pragma unroll
    for (int j = 0; j < 8; ++j) {
        c0[(size_t)j * ldc +  0] = acc0[j];
        c0[(size_t)j * ldc + 16] = acc1[j];
        c0[(size_t)j * ldc + 32] = acc2[j];
        c0[(size_t)j * ldc + 48] = acc3[j];
    }
}

// ---------------------------------------------------------------------------
// Elementwise / pack kernels
// ---------------------------------------------------------------------------
__global__ void k_pack_bf16(const float* __restrict__ s, bf16* __restrict__ d, size_t n) {
    size_t i = (size_t)blockIdx.x * blockDim.x + threadIdx.x;
    size_t stride = (size_t)gridDim.x * blockDim.x;
    for (; i < n; i += stride) d[i] = (bf16)s[i];
}
__global__ void k_zero_f32(float* __restrict__ p, size_t n) {
    size_t i = (size_t)blockIdx.x * blockDim.x + threadIdx.x;
    size_t stride = (size_t)gridDim.x * blockDim.x;
    for (; i < n; i += stride) p[i] = 0.0f;
}
__global__ void k_zero_bf16(bf16* __restrict__ p, size_t n) {
    size_t i = (size_t)blockIdx.x * blockDim.x + threadIdx.x;
    size_t stride = (size_t)gridDim.x * blockDim.x;
    for (; i < n; i += stride) p[i] = (bf16)0.0f;
}
// Wt[c][r] = W[r][c]   (W: R x C row-major -> Wt: C x R row-major)
__global__ void k_transpose_pack(const float* __restrict__ W, bf16* __restrict__ Wt,
                                 int R, int Ccols) {
    int idx = blockIdx.x * blockDim.x + threadIdx.x;
    if (idx >= R * Ccols) return;
    int c = idx / R, r = idx % R;
    Wt[idx] = (bf16)W[(size_t)r * Ccols + c];
}

// ---------------------------------------------------------------------------
// GAT phase 1: h@a_src1, h@a_dst1 (block-per-node dual dot product)
// ---------------------------------------------------------------------------
__global__ __launch_bounds__(256)
void k_matvec2(const float* __restrict__ h, const float* __restrict__ a1,
               const float* __restrict__ a2, float* __restrict__ o1, float* __restrict__ o2)
{
    __shared__ float s1[256], s2[256];
    const int i = blockIdx.x;
    const float* row = h + (size_t)i * D;
    float p1 = 0.0f, p2 = 0.0f;
    for (int d = threadIdx.x; d < D; d += 256) {
        float v = row[d];
        p1 += v * a1[d];
        p2 += v * a2[d];
    }
    s1[threadIdx.x] = p1; s2[threadIdx.x] = p2;
    __syncthreads();
    for (int s = 128; s > 0; s >>= 1) {
        if (threadIdx.x < (unsigned)s) {
            s1[threadIdx.x] += s1[threadIdx.x + s];
            s2[threadIdx.x] += s2[threadIdx.x + s];
        }
        __syncthreads();
    }
    if (threadIdx.x == 0) { o1[i] = s1[0]; o2[i] = s2[0]; }
}

__global__ void k_init_seg(unsigned* __restrict__ mxu, float* __restrict__ den) {
    int i = blockIdx.x * blockDim.x + threadIdx.x;
    if (i >= N) return;
    mxu[i] = fflip(-3.0e38f);
    den[i] = 0.0f;
}
__global__ void k_edge_e(const int* __restrict__ ei, const float* __restrict__ hsa,
                         const float* __restrict__ hda, float* __restrict__ e,
                         unsigned* __restrict__ mxu) {
    int j = blockIdx.x * blockDim.x + threadIdx.x;
    if (j >= E2) return;
    int s, t; edge_st(ei, j, s, t);
    float v = leaky(hsa[s] + hda[t]);
    e[j] = v;
    atomicMax(&mxu[t], fflip(v));
}
__global__ void k_edge_ex(const int* __restrict__ ei, const float* __restrict__ e,
                          const unsigned* __restrict__ mxu, float* __restrict__ ex,
                          float* __restrict__ den) {
    int j = blockIdx.x * blockDim.x + threadIdx.x;
    if (j >= E2) return;
    int s, t; edge_st(ei, j, s, t);
    float w = expf(e[j] - funflip(mxu[t]));
    ex[j] = w;
    atomicAdd(&den[t], w);
}
// x_gat accumulation: one block per edge, scatter-add alpha*h[src] into agg[tgt]
__global__ __launch_bounds__(256)
void k_aggregate(const int* __restrict__ ei, const float* __restrict__ ex,
                 const float* __restrict__ den, const float* __restrict__ h,
                 float* __restrict__ agg) {
    int j = blockIdx.x;
    int s, t; edge_st(ei, j, s, t);
    float alpha = ex[j] / den[t];
    const float* hr = h + (size_t)s * D;
    float* ar = agg + (size_t)t * D;
    for (int d = threadIdx.x; d < D; d += 256)
        atomicAdd(&ar[d], alpha * hr[d]);
}
// G = (x_gat + b1) * w2 packed to bf16 (A-operand of the H2 GEMM)
__global__ void k_finalizeG(const float* __restrict__ agg, const float* __restrict__ b1,
                            const float* __restrict__ w2, bf16* __restrict__ G) {
    size_t idx = (size_t)blockIdx.x * blockDim.x + threadIdx.x;
    size_t total = (size_t)N * D;
    size_t stride = (size_t)gridDim.x * blockDim.x;
    for (; idx < total; idx += stride) {
        int d = (int)(idx % D);
        G[idx] = (bf16)((agg[idx] + b1[d]) * w2[d]);
    }
}

// ---------------------------------------------------------------------------
// Reachability: A (bf16, row-major) and M1 = min(I + A^T, 1) built directly.
// ---------------------------------------------------------------------------
__global__ void k_build_adj(const int* __restrict__ ei, bf16* __restrict__ Abf,
                            float* __restrict__ M1f) {
    int j = blockIdx.x * blockDim.x + threadIdx.x;
    if (j >= E) return;
    int s = ei[j], t = ei[E + j];
    Abf[(size_t)s * N + t] = (bf16)1.0f;   // A[s][t] = 1
    M1f[(size_t)t * N + s] = 1.0f;         // A^T contribution
}
__global__ void k_diag(float* __restrict__ M1f) {
    int i = blockIdx.x * blockDim.x + threadIdx.x;
    if (i < N) M1f[(size_t)i * N + i] = 1.0f;
}
__global__ void k_hop_combine(const float* __restrict__ M1f, const float* __restrict__ Pf,
                              float* __restrict__ M2f, size_t n) {
    size_t i = (size_t)blockIdx.x * blockDim.x + threadIdx.x;
    size_t stride = (size_t)gridDim.x * blockDim.x;
    for (; i < n; i += stride) M2f[i] = fminf(M1f[i] + Pf[i], 1.0f);
}

// ---------------------------------------------------------------------------
// Per-row edge softmax: block per row; row state lives in LDS (30 KB/WG,
// well under the 320 KB/WGP budget).  Two passes over the edge list with
// DS atomics (flipped-uint max, float add).
// ---------------------------------------------------------------------------
__global__ __launch_bounds__(256)
void k_row_attn(const float* __restrict__ H2, const float* __restrict__ M2,
                const int* __restrict__ ei,
                const float* __restrict__ as2p, const float* __restrict__ ad2p,
                const float* __restrict__ b2p, float* __restrict__ score)
{
    __shared__ float    h2r[N];
    __shared__ float    mr[N];
    __shared__ unsigned msg[N];
    __shared__ float    dsg[N];
    __shared__ float    nsg[N];

    const int i = blockIdx.x;
    const float as2 = *as2p, ad2 = *ad2p, b2 = *b2p;

    for (int t = threadIdx.x; t < N; t += 256) {
        h2r[t] = H2[(size_t)i * N + t];
        mr[t]  = M2[(size_t)i * N + t];
        msg[t] = fflip(-3.0e38f);
        dsg[t] = 0.0f;
        nsg[t] = 0.0f;
    }
    __syncthreads();

    for (int j = threadIdx.x; j < E2; j += 256) {
        int s, t; edge_st(ei, j, s, t);
        bool valid = (mr[s] > 0.0f) && (mr[t] > 0.0f);
        float e2 = valid ? leaky(as2 * h2r[s] + ad2 * h2r[t]) : NEGV;
        atomicMax(&msg[t], fflip(e2));
    }
    __syncthreads();

    for (int j = threadIdx.x; j < E2; j += 256) {
        int s, t; edge_st(ei, j, s, t);
        float hs = h2r[s];
        bool valid = (mr[s] > 0.0f) && (mr[t] > 0.0f);
        float e2 = valid ? leaky(as2 * hs + ad2 * h2r[t]) : NEGV;
        float w = expf(e2 - funflip(msg[t]));
        atomicAdd(&dsg[t], w);
        atomicAdd(&nsg[t], w * hs);
    }
    __syncthreads();

    for (int t = threadIdx.x; t < N; t += 256)
        score[(size_t)i * N + t] = nsg[t] / fmaxf(dsg[t], 1e-12f) + b2;
}

// ---------------------------------------------------------------------------
// Top-k per row (stable-descending rank, ties broken by index as in
// jnp.argsort).  out = keep ? sigmoid(score) : 0 ; keep as 0/1 floats.
// ---------------------------------------------------------------------------
__global__ __launch_bounds__(256)
void k_topk(const float* __restrict__ score, const float* __restrict__ M2,
            float* __restrict__ outv, float* __restrict__ keepv)
{
    __shared__ float sc[N];     // masked scores
    __shared__ float sr[N];     // raw scores
    __shared__ unsigned char mb[N];
    __shared__ int cnt;

    const int i = blockIdx.x;
    if (threadIdx.x == 0) cnt = 0;
    __syncthreads();

    int local = 0;
    for (int t = threadIdx.x; t < N; t += 256) {
        float s = score[(size_t)i * N + t];
        bool m = M2[(size_t)i * N + t] > 0.0f;
        sr[t] = s;
        mb[t] = m ? 1 : 0;
        sc[t] = m ? s : NEGV;
        local += m ? 1 : 0;
    }
    atomicAdd(&cnt, local);
    __syncthreads();
    const int k = (cnt + 1) >> 1;    // ceil(0.5 * size)

    for (int t = threadIdx.x; t < N; t += 256) {
        bool keep = false;
        if (mb[t]) {
            const float st = sc[t];
            int rank = 0;
            for (int u = 0; u < N; ++u) {
                float su = sc[u];
                rank += (su > st) || (su == st && u < t);
            }
            keep = rank < k;
        }
        float s = sr[t];
        outv[(size_t)i * N + t]  = keep ? (1.0f / (1.0f + expf(-s))) : 0.0f;
        keepv[(size_t)i * N + t] = keep ? 1.0f : 0.0f;
    }
}

// ---------------------------------------------------------------------------
// Orchestration
// ---------------------------------------------------------------------------
extern "C" void kernel_launch(void* const* d_in, const int* in_sizes, int n_in,
                              void* d_out, int out_size, void* d_ws, size_t ws_size,
                              hipStream_t stream)
{
    const float* x      = (const float*)d_in[0];
    const float* xorig  = (const float*)d_in[1];
    const int*   ei     = (const int*)d_in[2];
    /* d_in[3] = batch, unused */
    const float* W1     = (const float*)d_in[4];
    const float* a_src1 = (const float*)d_in[5];
    const float* a_dst1 = (const float*)d_in[6];
    const float* b1     = (const float*)d_in[7];
    const float* w2v    = (const float*)d_in[8];
    const float* a_s2   = (const float*)d_in[9];
    const float* a_d2   = (const float*)d_in[10];
    const float* b2     = (const float*)d_in[11];
    (void)in_sizes; (void)n_in; (void)out_size; (void)ws_size;

    char* ws = (char*)d_ws;
    size_t off = 0;
    auto alloc = [&](size_t bytes) {
        void* p = ws + off;
        off = (off + bytes + 255) & ~(size_t)255;
        return p;
    };

    bf16*     xbf   = (bf16*)    alloc((size_t)N * HD * 2);
    bf16*     w1t   = (bf16*)    alloc((size_t)D * HD * 2);
    bf16*     xobf  = (bf16*)    alloc((size_t)N * D * 2);
    float*    h     = (float*)   alloc((size_t)N * D * 4);
    float*    hsa   = (float*)   alloc((size_t)N * 4);
    float*    hda   = (float*)   alloc((size_t)N * 4);
    float*    ebuf  = (float*)   alloc((size_t)E2 * 4);
    float*    exb   = (float*)   alloc((size_t)E2 * 4);
    unsigned* mxu   = (unsigned*)alloc((size_t)N * 4);
    float*    den   = (float*)   alloc((size_t)N * 4);
    float*    agg   = (float*)   alloc((size_t)N * D * 4);
    bf16*     Gbf   = (bf16*)    alloc((size_t)N * D * 2);
    bf16*     Abf   = (bf16*)    alloc((size_t)N * N * 2);
    float*    M1f   = (float*)   alloc((size_t)N * N * 4);
    bf16*     M1bf  = (bf16*)    alloc((size_t)N * N * 2);
    float*    Pf    = (float*)   alloc((size_t)N * N * 4);
    float*    M2f   = (float*)   alloc((size_t)N * N * 4);
    float*    scoreb= (float*)   alloc((size_t)N * N * 4);

    const dim3 b256(256);
    auto cdiv = [](size_t a, size_t b) { return (unsigned)((a + b - 1) / b); };

    // --- pack inputs for the matrix pipe ---
    k_pack_bf16<<<cdiv((size_t)N * HD, 256), b256, 0, stream>>>(x, xbf, (size_t)N * HD);
    k_pack_bf16<<<cdiv((size_t)N * D, 256),  b256, 0, stream>>>(xorig, xobf, (size_t)N * D);
    k_transpose_pack<<<cdiv((size_t)D * HD, 256), b256, 0, stream>>>(W1, w1t, HD, D);

    // --- h = x @ W1  (1536 x 128 x 1280) ---
    k_gemm_bf16<<<dim3(D / 64, N / 64), dim3(128), 0, stream>>>(xbf, HD, w1t, HD, h, D, HD);

    // --- GAT layer 1 edge softmax + aggregation ---
    k_matvec2<<<N, b256, 0, stream>>>(h, a_src1, a_dst1, hsa, hda);
    k_init_seg<<<cdiv(N, 256), b256, 0, stream>>>(mxu, den);
    k_edge_e<<<cdiv(E2, 256), b256, 0, stream>>>(ei, hsa, hda, ebuf, mxu);
    k_edge_ex<<<cdiv(E2, 256), b256, 0, stream>>>(ei, ebuf, mxu, exb, den);
    k_zero_f32<<<cdiv((size_t)N * D, 256), b256, 0, stream>>>(agg, (size_t)N * D);
    k_aggregate<<<E2, b256, 0, stream>>>(ei, exb, den, h, agg);
    k_finalizeG<<<cdiv((size_t)N * D, 256), b256, 0, stream>>>(agg, b1, w2v, Gbf);

    // --- 2-hop reachability: M1 = min(I + A^T,1) built directly; one GEMM for hop 2 ---
    k_zero_bf16<<<cdiv((size_t)N * N, 256), b256, 0, stream>>>(Abf, (size_t)N * N);
    k_zero_f32<<<cdiv((size_t)N * N, 256), b256, 0, stream>>>(M1f, (size_t)N * N);
    k_build_adj<<<cdiv(E, 256), b256, 0, stream>>>(ei, Abf, M1f);
    k_diag<<<cdiv(N, 256), b256, 0, stream>>>(M1f);
    k_pack_bf16<<<cdiv((size_t)N * N, 256), b256, 0, stream>>>(M1f, M1bf, (size_t)N * N);
    // P = M1 @ A^T  (B = A^T, so B^T = A row-major = Abf) — exact in bf16 (0/1 data)
    k_gemm_bf16<<<dim3(N / 64, N / 64), dim3(128), 0, stream>>>(M1bf, N, Abf, N, Pf, N, N);
    k_hop_combine<<<cdiv((size_t)N * N, 256), b256, 0, stream>>>(M1f, Pf, M2f, (size_t)N * N);

    // --- H2 = G @ x_orig^T  (1536 x 1280 x 1536); B^T = x_orig row-major ---
    k_gemm_bf16<<<dim3(N / 64, N / 64), dim3(128), 0, stream>>>(Gbf, D, xobf, D,
                                                                /*C=*/ (float*)Pf /*reuse? no*/, N, D);
    // NOTE: reuse of Pf would clobber nothing needed anymore, but keep clarity:
    // Pf is no longer needed after hop_combine, so it doubles as H2 storage.
    float* H2 = Pf;

    // --- per-row edge softmax scores ---
    k_row_attn<<<N, b256, 0, stream>>>(H2, M2f, ei, a_s2, a_d2, b2, scoreb);

    // --- top-k + sigmoid outputs ---
    float* outv  = (float*)d_out;
    float* keepv = outv + (size_t)N * N;
    k_topk<<<N, b256, 0, stream>>>(scoreb, M2f, outv, keepv);
}
